// ALiBiMultiHeadAttention_79680233275894
// MI455X (gfx1250) — compile-verified, matrix operand order checked by reading, not run on hardware
//
#include <hip/hip_runtime.h>
#include <hip/hip_bf16.h>

// ---------------------------------------------------------------------------
// ALiBi causal MHA for MI455X (gfx1250), wave32 + v_wmma_f32_16x16x32_f16.
// B=2, T=2048, D=1024, H=16, DK=64.
// Pipeline: cvt(f32->f16) -> QKV proj GEMM -> flash attention -> out proj GEMM
// All matmuls via WMMA f16 with f32 accumulation; register-prefetch double
// buffering overlaps global loads with WMMA compute.
// ---------------------------------------------------------------------------

typedef _Float16 f16;
typedef __attribute__((ext_vector_type(16))) _Float16 v16h;
typedef __attribute__((ext_vector_type(8)))  _Float16 v8h;
typedef __attribute__((ext_vector_type(4)))  _Float16 v4h;
typedef __attribute__((ext_vector_type(8)))  float    v8f;

#define BB 2
#define TT 2048
#define DD 1024
#define HH 16
#define DK 64

__device__ __forceinline__ v8f wmma16(v16h a, v16h b, v8f c) {
  // (neg_a, A, neg_b, B, c_mod, C, reuse_a, reuse_b)
  return __builtin_amdgcn_wmma_f32_16x16x32_f16(false, a, false, b, (short)0, c,
                                                false, false);
}

// A/B operand fragment loader (16-bit WMMA layout): lane holds row (M or N) =
// lane%16; per lane the 16 f16 elements are two contiguous 8-element chunks at
// K offsets kbase+8*half and kbase+16+8*half (half = lane/16).
__device__ __forceinline__ v16h ld_frag(const f16* rowp, int kbase, int hf) {
  v8h lo = *(const v8h*)(rowp + kbase + 8 * hf);
  v8h hi = *(const v8h*)(rowp + kbase + 16 + 8 * hf);
  v16h r;
#pragma unroll
  for (int i = 0; i < 8; ++i) { r[i] = lo[i]; r[8 + i] = hi[i]; }
  return r;
}

// ---------------------------------------------------------------------------
// f32 -> f16 conversion, vectorized x8 (two float4 loads -> one 16B store).
// ---------------------------------------------------------------------------
__global__ __launch_bounds__(256) void cvt_kernel(const float* __restrict__ src,
                                                  f16* __restrict__ dst, int n8) {
  int i = blockIdx.x * blockDim.x + threadIdx.x;
  if (i < n8) {
    float4 f0 = ((const float4*)src)[2 * i + 0];
    float4 f1 = ((const float4*)src)[2 * i + 1];
    v8h h;
    h[0] = (f16)f0.x; h[1] = (f16)f0.y; h[2] = (f16)f0.z; h[3] = (f16)f0.w;
    h[4] = (f16)f1.x; h[5] = (f16)f1.y; h[6] = (f16)f1.z; h[7] = (f16)f1.w;
    ((v8h*)dst)[i] = h;
  }
}

// ---------------------------------------------------------------------------
// Unified GEMM: acc = A16 @ W16^T  (both read row-major along K).
// Block tile 128x128, BK=32, 256 threads = 8 waves (4x2), wave tile 32x64.
// Register-prefetch double buffering: next K-step's global loads are issued
// before the current step's WMMAs so they overlap.
// MODE 0: Out f16 head-split [B,H,T,DK], value * scale   (QKV projections)
// MODE 1: Out f32 flat [B*T, D] + bias                    (output projection)
// ---------------------------------------------------------------------------
template <int MODE>
__global__ __launch_bounds__(256) void gemm_kernel(
    const f16* __restrict__ A16,   // [B*T, D]
    const f16* __restrict__ W16,   // [D, D], row n = weights of out col n
    f16* __restrict__ out16,       // MODE 0
    float* __restrict__ out32,     // MODE 1
    const float* __restrict__ bias,// MODE 1
    float scale) {
  __shared__ f16 As[128 * 40];
  __shared__ f16 Bs[128 * 40];
  const int tid = threadIdx.x;
  const int lane = tid & 31, wave = tid >> 5;
  const int hf = lane >> 4, ln16 = lane & 15;
  const int wm = wave & 3, wn = wave >> 2;
  const int m0 = blockIdx.x * 128;
  const int n0 = blockIdx.y * 128;

  // Per-thread staging chunks: c = tid + 256*jj; row = c>>2, ch = c&3.
  const int row0 = tid >> 2, ch0 = tid & 3;
  const int row1 = (tid + 256) >> 2, ch1 = (tid + 256) & 3;
  const f16* agp0 = A16 + (size_t)(m0 + row0) * DD + ch0 * 8;
  const f16* agp1 = A16 + (size_t)(m0 + row1) * DD + ch1 * 8;
  const f16* bgp0 = W16 + (size_t)(n0 + row0) * DD + ch0 * 8;
  const f16* bgp1 = W16 + (size_t)(n0 + row1) * DD + ch1 * 8;
  f16* asp0 = As + row0 * 40 + ch0 * 8;
  f16* asp1 = As + row1 * 40 + ch1 * 8;
  f16* bsp0 = Bs + row0 * 40 + ch0 * 8;
  f16* bsp1 = Bs + row1 * 40 + ch1 * 8;

  v8f acc[2][4] = {};
  v8h ar0, ar1, br0, br1;

  // Preload K-step 0.
  ar0 = *(const v8h*)(agp0);
  ar1 = *(const v8h*)(agp1);
  br0 = *(const v8h*)(bgp0);
  br1 = *(const v8h*)(bgp1);

  for (int it = 0; it < DD / 32; ++it) {
    __syncthreads();  // previous iteration's readers are done with LDS
    *(v8h*)(asp0) = ar0;
    *(v8h*)(asp1) = ar1;
    *(v8h*)(bsp0) = br0;
    *(v8h*)(bsp1) = br1;
    __syncthreads();  // staged tile visible to all waves

    // Prefetch next K-step (overlaps with the 8 WMMAs below).
    if (it + 1 < DD / 32) {
      int k0 = (it + 1) * 32;
      ar0 = *(const v8h*)(agp0 + k0);
      ar1 = *(const v8h*)(agp1 + k0);
      br0 = *(const v8h*)(bgp0 + k0);
      br1 = *(const v8h*)(bgp1 + k0);
    }

    v16h af[2], bf[4];
#pragma unroll
    for (int i = 0; i < 2; ++i)
      af[i] = ld_frag(As + (wm * 32 + i * 16 + ln16) * 40, 0, hf);
#pragma unroll
    for (int j = 0; j < 4; ++j)
      bf[j] = ld_frag(Bs + (wn * 64 + j * 16 + ln16) * 40, 0, hf);
#pragma unroll
    for (int i = 0; i < 2; ++i)
#pragma unroll
      for (int j = 0; j < 4; ++j) acc[i][j] = wmma16(af[i], bf[j], acc[i][j]);
  }

  // Epilogue. C layout: element (VGPR v, lane) = row v+8*hf, col ln16.
#pragma unroll
  for (int i = 0; i < 2; ++i) {
#pragma unroll
    for (int j = 0; j < 4; ++j) {
      int n = n0 + wn * 64 + j * 16 + ln16;
      if (MODE == 0) {
        int h = n >> 6, d = n & 63;
#pragma unroll
        for (int v = 0; v < 8; ++v) {
          int m = m0 + wm * 32 + i * 16 + v + 8 * hf;
          int b = m >> 11, t = m & (TT - 1);
          out16[(((size_t)(b * HH + h)) * TT + t) * DK + d] =
              (f16)(acc[i][j][v] * scale);
        }
      } else {
        float bn = bias[n];
#pragma unroll
        for (int v = 0; v < 8; ++v) {
          int m = m0 + wm * 32 + i * 16 + v + 8 * hf;
          out32[(size_t)m * DD + n] = acc[i][j][v] + bn;
        }
      }
    }
  }
}

// ---------------------------------------------------------------------------
// Flash attention with analytic ALiBi bias + causal mask.
// grid = (B*H, T/64); block = 128 threads = 4 waves; wave owns 16 q rows.
// K/V tiles register-prefetched one iteration ahead.
// ---------------------------------------------------------------------------
__global__ __launch_bounds__(128) void attn_kernel(
    const f16* __restrict__ Qh,    // [B,H,T,DK] (pre-scaled by 1/8)
    const f16* __restrict__ Kh,    // [B,H,T,DK]
    const f16* __restrict__ Vh,    // [B,H,T,DK]
    f16* __restrict__ AttnOut) {   // [B*T, D] f16
  __shared__ f16 Ks[64 * 72];       // K tile, row-major [key][dk]
  __shared__ f16 Vt[64 * 72];       // V tile transposed [dk][key]
  __shared__ f16 Pl[4 * 16 * 72];   // per-wave P relayout buffer

  const int bh = blockIdx.x;
  const int b = bh >> 4, h = bh & 15;
  const int q0 = blockIdx.y * 64;
  const int tid = threadIdx.x, lane = tid & 31, wave = tid >> 5;
  const int hf = lane >> 4, ln16 = lane & 15;
  const float slope = __builtin_exp2f(-0.5f * (float)(h + 1));

  const f16* Qb = Qh + (size_t)bh * TT * DK;
  const f16* Kb = Kh + (size_t)bh * TT * DK;
  const f16* Vb = Vh + (size_t)bh * TT * DK;

  // Q fragments (A layout), kept in registers for the whole block.
  v16h qf[2];
  {
    const f16* qr = Qb + (size_t)(q0 + wave * 16 + ln16) * DK;
    qf[0] = ld_frag(qr, 0, hf);
    qf[1] = ld_frag(qr, 32, hf);
  }

  v8f O[4] = {};
  float mprev[8], lsum[8];
#pragma unroll
  for (int v = 0; v < 8; ++v) { mprev[v] = -3.0e38f; lsum[v] = 0.f; }
  f16* Pw = Pl + wave * 16 * 72;

  // Staging chunk mapping: c = tid + 128*jj; row = c>>3 (key), ch = c&7.
  const int ktmax = q0 >> 6;  // inclusive: tiles up to the diagonal

  v8h kr[4], vr[4];
#pragma unroll
  for (int jj = 0; jj < 4; ++jj) {
    int c = tid + 128 * jj;
    int row = c >> 3, ch = c & 7;
    kr[jj] = *(const v8h*)(Kb + (size_t)row * DK + ch * 8);
    vr[jj] = *(const v8h*)(Vb + (size_t)row * DK + ch * 8);
  }

  for (int kt = 0; kt <= ktmax; ++kt) {
    __syncthreads();  // everyone done reading previous K/V tile
#pragma unroll
    for (int jj = 0; jj < 4; ++jj) {
      int c = tid + 128 * jj;
      int row = c >> 3, ch = c & 7;
      *(v8h*)(Ks + row * 72 + ch * 8) = kr[jj];
#pragma unroll
      for (int e = 0; e < 8; ++e) Vt[(ch * 8 + e) * 72 + row] = vr[jj][e];
    }
    __syncthreads();

    // Prefetch next tile (overlaps with WMMAs + softmax below).
    if (kt < ktmax) {
#pragma unroll
      for (int jj = 0; jj < 4; ++jj) {
        int c = tid + 128 * jj;
        int row = c >> 3, ch = c & 7;
        kr[jj] = *(const v8h*)(Kb + (size_t)((kt + 1) * 64 + row) * DK + ch * 8);
        vr[jj] = *(const v8h*)(Vb + (size_t)((kt + 1) * 64 + row) * DK + ch * 8);
      }
    }

    // S = Q @ K^T  (pre-scaled): 8 WMMAs
    v8f st[4] = {};
#pragma unroll
    for (int c = 0; c < 2; ++c)
#pragma unroll
      for (int j = 0; j < 4; ++j) {
        v16h bf = ld_frag(Ks + (j * 16 + ln16) * 72, 32 * c, hf);
        st[j] = wmma16(qf[c], bf, st[j]);
      }

    // ALiBi bias + causal mask + online softmax (rows span 16 lanes).
    const int rowbase = q0 + wave * 16 + 8 * hf;
#pragma unroll
    for (int v = 0; v < 8; ++v) {
      int r = rowbase + v;
      float mcur = -3.0e38f;
#pragma unroll
      for (int j = 0; j < 4; ++j) {
        int cabs = kt * 64 + j * 16 + ln16;
        float s = st[j][v];
        s = (cabs <= r) ? (s + slope * (float)(cabs - r)) : -3.0e38f;
        st[j][v] = s;
        mcur = fmaxf(mcur, s);
      }
#pragma unroll
      for (int off = 8; off >= 1; off >>= 1)
        mcur = fmaxf(mcur, __shfl_xor(mcur, off, 16));
      float mnew = fmaxf(mprev[v], mcur);
      float alpha = __expf(mprev[v] - mnew);
      float rsum = 0.f;
#pragma unroll
      for (int j = 0; j < 4; ++j) {
        float p = __expf(st[j][v] - mnew);
        st[j][v] = p;
        rsum += p;
      }
#pragma unroll
      for (int off = 8; off >= 1; off >>= 1)
        rsum += __shfl_xor(rsum, off, 16);
      lsum[v] = lsum[v] * alpha + rsum;
      mprev[v] = mnew;
#pragma unroll
      for (int j = 0; j < 4; ++j) O[j][v] *= alpha;
    }

    // Re-layout P (C layout -> A layout) through per-wave LDS.
    // DS ops are in-order within a wave: no barrier needed.
#pragma unroll
    for (int j = 0; j < 4; ++j)
#pragma unroll
      for (int v = 0; v < 8; ++v)
        Pw[(v + 8 * hf) * 72 + j * 16 + ln16] = (f16)st[j][v];

    // O += P @ V : 8 WMMAs (B operand = rows of transposed V tile)
#pragma unroll
    for (int c = 0; c < 2; ++c) {
      v16h pa = ld_frag(Pw + ln16 * 72, 32 * c, hf);
#pragma unroll
      for (int j = 0; j < 4; ++j) {
        v16h bf = ld_frag(Vt + (j * 16 + ln16) * 72, 32 * c, hf);
        O[j] = wmma16(pa, bf, O[j]);
      }
    }
  }

  // Epilogue: O/l -> f16, merge heads back to [B*T, D].
#pragma unroll
  for (int v = 0; v < 8; ++v) {
    float inv = 1.0f / lsum[v];
    int t = q0 + wave * 16 + v + 8 * hf;
    size_t rowoff = ((size_t)b * TT + t) * DD + h * DK;
#pragma unroll
    for (int j = 0; j < 4; ++j)
      AttnOut[rowoff + j * 16 + ln16] = (f16)(O[j][v] * inv);
  }
}

// ---------------------------------------------------------------------------
// Host orchestration.
// ---------------------------------------------------------------------------
extern "C" void kernel_launch(void* const* d_in, const int* in_sizes, int n_in,
                              void* d_out, int out_size, void* d_ws,
                              size_t ws_size, hipStream_t stream) {
  const float* q  = (const float*)d_in[0];
  const float* k  = (const float*)d_in[1];
  const float* v  = (const float*)d_in[2];
  // d_in[3] = alibi_bias: computed analytically in-kernel (identical formula).
  const float* Wq = (const float*)d_in[4];
  const float* Wk = (const float*)d_in[5];
  const float* Wv = (const float*)d_in[6];
  const float* Wo = (const float*)d_in[7];
  const float* bo = (const float*)d_in[8];
  float* out = (float*)d_out;

  // Workspace (all f16): 4 weights (2MB) + 3 activations (8MB) +
  // Qh/Kh/Vh (8MB) + attn-out (8MB) = 64MB total.
  char* ws = (char*)d_ws;
  const size_t WSZ = (size_t)DD * DD * sizeof(f16);       // 2 MB
  const size_t ASZ = (size_t)BB * TT * DD * sizeof(f16);  // 8 MB
  f16* W16q = (f16*)(ws);                 ws += WSZ;
  f16* W16k = (f16*)(ws);                 ws += WSZ;
  f16* W16v = (f16*)(ws);                 ws += WSZ;
  f16* W16o = (f16*)(ws);                 ws += WSZ;
  f16* X16q = (f16*)(ws);                 ws += ASZ;
  f16* X16k = (f16*)(ws);                 ws += ASZ;
  f16* X16v = (f16*)(ws);                 ws += ASZ;
  f16* Qh   = (f16*)(ws);                 ws += ASZ;
  f16* Kh   = (f16*)(ws);                 ws += ASZ;
  f16* Vh   = (f16*)(ws);                 ws += ASZ;
  f16* At16 = (f16*)(ws);                 ws += ASZ;

  // 1) f32 -> f16: weights and activations
  {
    int n8w = DD * DD / 8;
    int grdw = (n8w + 255) / 256;
    cvt_kernel<<<grdw, 256, 0, stream>>>(Wq, W16q, n8w);
    cvt_kernel<<<grdw, 256, 0, stream>>>(Wk, W16k, n8w);
    cvt_kernel<<<grdw, 256, 0, stream>>>(Wv, W16v, n8w);
    cvt_kernel<<<grdw, 256, 0, stream>>>(Wo, W16o, n8w);
    int n8a = BB * TT * DD / 8;
    int grda = (n8a + 255) / 256;
    cvt_kernel<<<grda, 256, 0, stream>>>(q, X16q, n8a);
    cvt_kernel<<<grda, 256, 0, stream>>>(k, X16k, n8a);
    cvt_kernel<<<grda, 256, 0, stream>>>(v, X16v, n8a);
  }

  // 2) QKV projections (scale 1/sqrt(DK) folded into Q)
  {
    dim3 grid(BB * TT / 128, DD / 128);
    gemm_kernel<0><<<grid, 256, 0, stream>>>(X16q, W16q, Qh, nullptr, nullptr, 0.125f);
    gemm_kernel<0><<<grid, 256, 0, stream>>>(X16k, W16k, Kh, nullptr, nullptr, 1.0f);
    gemm_kernel<0><<<grid, 256, 0, stream>>>(X16v, W16v, Vh, nullptr, nullptr, 1.0f);
  }

  // 3) Flash attention with ALiBi + causal mask
  {
    dim3 grid(BB * HH, TT / 64);
    attn_kernel<<<grid, 128, 0, stream>>>(Qh, Kh, Vh, At16);
  }

  // 4) Output projection + bias
  {
    dim3 grid(BB * TT / 128, DD / 128);
    gemm_kernel<1><<<grid, 256, 0, stream>>>(At16, W16o, nullptr, out, bo, 1.0f);
  }
}
